// GCN_model_45896020525326
// MI455X (gfx1250) — matrix operand order repaired and checked by reading, not compiled, not run on hardware
//
#include <hip/hip_runtime.h>
#include <hip/hip_bf16.h>

#define NN  50000
#define NE  800000
#define DIN 256
#define HID 512

typedef __attribute__((ext_vector_type(16))) __bf16 v16bf;
typedef __attribute__((ext_vector_type(8)))  __bf16 v8bf;
typedef __attribute__((ext_vector_type(8)))  float  v8f;

// ---------------------------------------------------------------------------
// f32 -> bf16 elementwise convert
// ---------------------------------------------------------------------------
__global__ void f32_to_bf16(const float* __restrict__ in, __bf16* __restrict__ out, long total) {
    long t = (long)blockIdx.x * blockDim.x + threadIdx.x;
    if (t >= total) return;
    out[t] = (__bf16)in[t];
}

// ---------------------------------------------------------------------------
// Pack weight W[K x N] (row-major f32) into WMMA B-fragment layout (bf16).
// Packed element index = ((nt*(K/32)+kb)*32 + lane)*16 + i
//   where lane's element i holds B[kb*32 + (lane<16?0:8) + (i&7) + (i>=8?16:0), nt*16 + lane%16]
// so each lane's 16 bf16 for one (nt,kb) tile are contiguous (32 B).
// ---------------------------------------------------------------------------
__global__ void pack_weight_bf16(const float* __restrict__ W, __bf16* __restrict__ P,
                                 int K, int N) {
    int tid = blockIdx.x * blockDim.x + threadIdx.x;
    if (tid >= K * N) return;
    const int KB   = K >> 5;
    const int i    = tid & 15;
    const int lane = (tid >> 4) & 31;
    const int rem  = tid >> 9;
    const int kb   = rem % KB;
    const int nt   = rem / KB;
    const int k    = (kb << 5) + ((lane >> 4) << 3) + (i & 7) + ((i >> 3) << 4);
    const int n    = (nt << 4) + (lane & 15);
    P[tid] = (__bf16)W[(size_t)k * N + n];
}

// ---------------------------------------------------------------------------
// WMMA GEMM: out[M,N] = X[M,K](bf16) @ Wp(packed bf16), f32 accumulate.
// One wave computes a 16x64 strip (4 n-tiles reuse one A fragment).
// outBf16 selects bf16 or f32 output buffer.
// ---------------------------------------------------------------------------
__global__ void __launch_bounds__(256)
gemm_bf16_wmma(const __bf16* __restrict__ X, const __bf16* __restrict__ Wp,
               float* __restrict__ outF, __bf16* __restrict__ outB,
               int M, int K, int N, int outBf16)
{
    const int lane  = threadIdx.x & 31;
    const int wave  = threadIdx.x >> 5;
    const int mtile = blockIdx.x * 8 + wave;
    if (mtile * 16 >= M) return;                 // wave-uniform: EXEC stays all-1s
    const int    ntbase   = blockIdx.y << 2;     // 4 n-tiles of 16
    const int    arow     = mtile * 16 + (lane & 15);
    const int    hi       = lane >> 4;           // lane group 0/1
    const int    KB       = K >> 5;
    const size_t ntstride = (size_t)KB << 9;     // elements per packed n-tile

    v8f c0 = {}; v8f c1 = {}; v8f c2 = {}; v8f c3 = {};
    const __bf16* xrow = X + (size_t)arow * K;
    const __bf16* wb   = Wp + (size_t)ntbase * ntstride + ((size_t)lane << 4);

    for (int kb = 0; kb < KB; ++kb) {
        const int ka = (kb << 5) + hi * 8;
        v8bf a0 = *(const v8bf*)(xrow + ka);          // K = ka..ka+7
        v8bf a1 = *(const v8bf*)(xrow + ka + 16);     // K = ka+16..ka+23
        v16bf a = __builtin_shufflevector(a0, a1, 0,1,2,3,4,5,6,7,8,9,10,11,12,13,14,15);

        const __bf16* bp = wb + ((size_t)kb << 9);
        v16bf b0 = *(const v16bf*)(bp);
        v16bf b1 = *(const v16bf*)(bp + ntstride);
        v16bf b2 = *(const v16bf*)(bp + 2 * ntstride);
        v16bf b3 = *(const v16bf*)(bp + 3 * ntstride);

        c0 = __builtin_amdgcn_wmma_f32_16x16x32_bf16(false, a, false, b0, (short)0, c0, false, false);
        c1 = __builtin_amdgcn_wmma_f32_16x16x32_bf16(false, a, false, b1, (short)0, c1, false, false);
        c2 = __builtin_amdgcn_wmma_f32_16x16x32_bf16(false, a, false, b2, (short)0, c2, false, false);
        c3 = __builtin_amdgcn_wmma_f32_16x16x32_bf16(false, a, false, b3, (short)0, c3, false, false);
    }

    const int orow = mtile * 16 + hi * 8;
    const int col0 = (ntbase << 4) + (lane & 15);
    if (outBf16) {
#define GCN_ST_B(cj, j) { const int col = col0 + (j) * 16; \
        _Pragma("unroll") for (int r = 0; r < 8; ++r) \
            outB[(size_t)(orow + r) * N + col] = (__bf16)cj[r]; }
        GCN_ST_B(c0, 0) GCN_ST_B(c1, 1) GCN_ST_B(c2, 2) GCN_ST_B(c3, 3)
#undef GCN_ST_B
    } else {
#define GCN_ST_F(cj, j) { const int col = col0 + (j) * 16; \
        _Pragma("unroll") for (int r = 0; r < 8; ++r) \
            outF[(size_t)(orow + r) * N + col] = cj[r]; }
        GCN_ST_F(c0, 0) GCN_ST_F(c1, 1) GCN_ST_F(c2, 2) GCN_ST_F(c3, 3)
#undef GCN_ST_F
    }
}

// ---------------------------------------------------------------------------
// Degree / normalization precompute (self-loop folded into deg init = 1)
// ---------------------------------------------------------------------------
__global__ void deg_init(float* __restrict__ deg) {
    int n = blockIdx.x * blockDim.x + threadIdx.x;
    if (n < NN) deg[n] = 1.0f;                    // self loop
}
__global__ void deg_accum(const int* __restrict__ dst, float* __restrict__ deg) {
    int e = blockIdx.x * blockDim.x + threadIdx.x;
    if (e < NE) atomicAdd(&deg[dst[e]], 1.0f);
}
__global__ void make_dis(const float* __restrict__ deg, float* __restrict__ dis,
                         float* __restrict__ selfnorm) {
    int n = blockIdx.x * blockDim.x + threadIdx.x;
    if (n >= NN) return;
    float r = rsqrtf(deg[n]);
    dis[n] = r;
    selfnorm[n] = r * r;
}
__global__ void make_norm(const int* __restrict__ src, const int* __restrict__ dst,
                          const float* __restrict__ dis, float* __restrict__ nrm) {
    int e = blockIdx.x * blockDim.x + threadIdx.x;
    if (e >= NE) return;
    nrm[e] = dis[src[e]] * dis[dst[e]];
}

// ---------------------------------------------------------------------------
// AGG[n,c] = bias[c] + selfnorm[n] * H[n,c]   (C == HID, fixed)
// ---------------------------------------------------------------------------
__global__ void agg_init(const __bf16* __restrict__ H, const float* __restrict__ bias,
                         const float* __restrict__ selfnorm, float* __restrict__ AGG) {
    long t = (long)blockIdx.x * blockDim.x + threadIdx.x;
    if (t >= (long)NN * HID) return;
    const int n = (int)(t >> 9);
    const int c = (int)(t & (HID - 1));
    AGG[t] = bias[c] + (float)H[t] * selfnorm[n];
}

// ---------------------------------------------------------------------------
// AGG[dst[e], :] += norm[e] * H[src[e], :]  — AGG (102 MB) is L2-resident,
// atomics stay on-chip. blockDim = (HID/4, 2): 2 edges per block, 4 ch/thread.
// ---------------------------------------------------------------------------
__global__ void __launch_bounds__(256)
scatter_edges(const __bf16* __restrict__ H, float* __restrict__ AGG,
              const int* __restrict__ src, const int* __restrict__ dst,
              const float* __restrict__ nrm)
{
    const int e = blockIdx.x * 2 + threadIdx.y;
    if (e >= NE) return;
    const int   s  = src[e];
    const int   d  = dst[e];
    const float w  = nrm[e];
    const int   c4 = threadIdx.x << 2;
    const __bf16* hp = H + (size_t)s * HID + c4;
    float* ap = AGG + (size_t)d * HID + c4;
    atomicAdd(ap + 0, (float)hp[0] * w);
    atomicAdd(ap + 1, (float)hp[1] * w);
    atomicAdd(ap + 2, (float)hp[2] * w);
    atomicAdd(ap + 3, (float)hp[3] * w);
}

// ---------------------------------------------------------------------------
// out = bf16(relu(in [+ bias]))
// ---------------------------------------------------------------------------
__global__ void bias_relu_bf16(const float* __restrict__ in, const float* __restrict__ bias,
                               int useBias, __bf16* __restrict__ out, int C, long total) {
    long t = (long)blockIdx.x * blockDim.x + threadIdx.x;
    if (t >= total) return;
    float v = in[t];
    if (useBias) v += bias[(int)(t & (C - 1))];
    out[t] = (__bf16)(v > 0.0f ? v : 0.0f);
}

// ---------------------------------------------------------------------------
// Final 256->2 linear + softmax; writes logits then probs (concatenated).
// ---------------------------------------------------------------------------
__global__ void head_kernel(const __bf16* __restrict__ H, const float* __restrict__ W,
                            const float* __restrict__ b, float* __restrict__ out) {
    int n = blockIdx.x * blockDim.x + threadIdx.x;
    if (n >= NN) return;
    const __bf16* h = H + (size_t)n * (HID / 2);
    float a0 = b[0], a1 = b[1];
    for (int c = 0; c < HID / 2; ++c) {
        float v = (float)h[c];
        a0 = fmaf(v, W[c * 2 + 0], a0);
        a1 = fmaf(v, W[c * 2 + 1], a1);
    }
    out[(size_t)n * 2 + 0] = a0;
    out[(size_t)n * 2 + 1] = a1;
    float m  = fmaxf(a0, a1);
    float e0 = __expf(a0 - m), e1 = __expf(a1 - m);
    float inv = 1.0f / (e0 + e1);
    out[(size_t)2 * NN + n * 2 + 0] = e0 * inv;
    out[(size_t)2 * NN + n * 2 + 1] = e1 * inv;
}

// ---------------------------------------------------------------------------
extern "C" void kernel_launch(void* const* d_in, const int* in_sizes, int n_in,
                              void* d_out, int out_size, void* d_ws, size_t ws_size,
                              hipStream_t stream) {
    (void)in_sizes; (void)n_in; (void)out_size; (void)ws_size;

    const float* x    = (const float*)d_in[0];
    const int*   ei   = (const int*)d_in[1];
    const int*   esrc = ei;
    const int*   edst = ei + NE;
    const float* W1  = (const float*)d_in[2];  const float* b1  = (const float*)d_in[3];
    const float* W2  = (const float*)d_in[4];  const float* b2  = (const float*)d_in[5];
    const float* W3  = (const float*)d_in[6];  const float* b3  = (const float*)d_in[7];
    const float* W4  = (const float*)d_in[8];  const float* b4  = (const float*)d_in[9];
    const float* Wl1 = (const float*)d_in[10]; const float* bl1 = (const float*)d_in[11];
    const float* Wl2 = (const float*)d_in[12]; const float* bl2 = (const float*)d_in[13];
    const float* Wl3 = (const float*)d_in[14]; const float* bl3 = (const float*)d_in[15];
    float* out = (float*)d_out;

    // ---- workspace carve ----
    char* p = (char*)d_ws;
    auto carve = [&](size_t bytes) {
        void* r = (void*)p;
        p += (bytes + 255) & ~(size_t)255;
        return r;
    };
    __bf16* W1p  = (__bf16*)carve((size_t)DIN * HID * 2);
    __bf16* W2p  = (__bf16*)carve((size_t)HID * HID * 2);
    __bf16* W3p  = (__bf16*)carve((size_t)HID * HID * 2);
    __bf16* W4p  = (__bf16*)carve((size_t)HID * HID * 2);
    __bf16* Wl1p = (__bf16*)carve((size_t)HID * HID * 2);
    __bf16* Wl2p = (__bf16*)carve((size_t)HID * (HID / 2) * 2);
    __bf16* Xb   = (__bf16*)carve((size_t)NN * HID * 2);   // activations (bf16)
    __bf16* Hb   = (__bf16*)carve((size_t)NN * HID * 2);   // GEMM output (bf16)
    float*  AGG  = (float*)carve((size_t)NN * HID * 4);    // f32 accumulator (L2 resident)
    float*  deg  = (float*)carve((size_t)NN * 4);
    float*  dis  = (float*)carve((size_t)NN * 4);
    float*  sn   = (float*)carve((size_t)NN * 4);
    float*  nrm  = (float*)carve((size_t)NE * 4);

    const int TPB = 256;
    const int MT  = NN / 16;  // 3125 m-tiles
    dim3 gblock(256);

    // ---- pack weights to WMMA B layout (bf16) ----
    pack_weight_bf16<<<(DIN * HID + TPB - 1) / TPB, TPB, 0, stream>>>(W1, W1p, DIN, HID);
    pack_weight_bf16<<<(HID * HID + TPB - 1) / TPB, TPB, 0, stream>>>(W2, W2p, HID, HID);
    pack_weight_bf16<<<(HID * HID + TPB - 1) / TPB, TPB, 0, stream>>>(W3, W3p, HID, HID);
    pack_weight_bf16<<<(HID * HID + TPB - 1) / TPB, TPB, 0, stream>>>(W4, W4p, HID, HID);
    pack_weight_bf16<<<(HID * HID + TPB - 1) / TPB, TPB, 0, stream>>>(Wl1, Wl1p, HID, HID);
    pack_weight_bf16<<<(HID * (HID / 2) + TPB - 1) / TPB, TPB, 0, stream>>>(Wl2, Wl2p, HID, HID / 2);

    // ---- input to bf16 ----
    {
        long tot = (long)NN * DIN;
        f32_to_bf16<<<(unsigned)((tot + TPB - 1) / TPB), TPB, 0, stream>>>(x, Xb, tot);
    }

    // ---- degree / norm precompute ----
    deg_init<<<(NN + TPB - 1) / TPB, TPB, 0, stream>>>(deg);
    deg_accum<<<(NE + TPB - 1) / TPB, TPB, 0, stream>>>(edst, deg);
    make_dis<<<(NN + TPB - 1) / TPB, TPB, 0, stream>>>(deg, dis, sn);
    make_norm<<<(NE + TPB - 1) / TPB, TPB, 0, stream>>>(esrc, edst, dis, nrm);

    // ---- 4 GCN conv layers ----
    auto conv = [&](const __bf16* Wp, const float* bias, int K) {
        dim3 ggrid((MT + 7) / 8, HID / 64);
        gemm_bf16_wmma<<<ggrid, gblock, 0, stream>>>(Xb, Wp, nullptr, Hb, NN, K, HID, 1);
        long tot = (long)NN * HID;
        agg_init<<<(unsigned)((tot + TPB - 1) / TPB), TPB, 0, stream>>>(Hb, bias, sn, AGG);
        dim3 sgrid((NE + 1) / 2);
        dim3 sblock(HID / 4, 2);
        scatter_edges<<<sgrid, sblock, 0, stream>>>(Hb, AGG, esrc, edst, nrm);
        bias_relu_bf16<<<(unsigned)((tot + TPB - 1) / TPB), TPB, 0, stream>>>(
            AGG, nullptr, 0, Xb, HID, tot);
    };
    conv(W1p, b1, DIN);
    conv(W2p, b2, HID);
    conv(W3p, b3, HID);
    conv(W4p, b4, HID);

    // ---- lin1: 512 -> 512, relu ----
    {
        dim3 ggrid((MT + 7) / 8, HID / 64);
        gemm_bf16_wmma<<<ggrid, gblock, 0, stream>>>(Xb, Wl1p, AGG, nullptr, NN, HID, HID, 0);
        long tot = (long)NN * HID;
        bias_relu_bf16<<<(unsigned)((tot + TPB - 1) / TPB), TPB, 0, stream>>>(
            AGG, bl1, 1, Xb, HID, tot);
    }
    // ---- lin2: 512 -> 256, relu ----
    {
        dim3 ggrid((MT + 7) / 8, (HID / 2) / 64);
        gemm_bf16_wmma<<<ggrid, gblock, 0, stream>>>(Xb, Wl2p, AGG, nullptr, NN, HID, HID / 2, 0);
        long tot = (long)NN * (HID / 2);
        bias_relu_bf16<<<(unsigned)((tot + TPB - 1) / TPB), TPB, 0, stream>>>(
            AGG, bl2, 1, Xb, HID / 2, tot);
    }
    // ---- head: 256 -> 2 + softmax ----
    head_kernel<<<(NN + TPB - 1) / TPB, TPB, 0, stream>>>(Xb, Wl3, bl3, out);
}
